// TemporalGatedCrossModalMetaAdapter_12618613916093
// MI455X (gfx1250) — matrix operand order repaired and checked by reading, not compiled, not run on hardware
//
#include <hip/hip_runtime.h>

// ---------------------------------------------------------------------------
// TemporalGatedCrossModalMetaAdapter, restructured for MI455X (gfx1250).
//
// C=2048, KT=64, D=1024, De=128, Df=256, H=8, dh=128, PH=1024, X=D+De=1152.
//
// Algebraic refactor: never materialize K/V [C,KT,D].
//   Qt[c,h,x]   = sum_j W1[h*128+j, x] * Q[c, h*128+j]      (per-head GEMM)
//   scores[k,h] = Fp[k,:] . Qt[c,h,:] / sqrt(dh)            (WMMA, streams Fp)
//   aF[c,h,x]   = sum_k attn[h,k] * Fp[k,x]                 (WMMA, streams Fp)
//   A[c,d]      = Wv[d,:] . aF[c, d/128, :] + alpha*fCp[c,d]
// HBM floor: read F+Et once = 576 MB  =>  ~25 us @ 23.3 TB/s (memory-bound).
// All GEMMs in bf16 WMMA (v_wmma_f32_16x16x32_bf16), M-tile = 16 chunks.
// Et[c] tile is DMA'd into LDS with the Tensor Data Mover
// (tensor_load_to_lds + s_wait_tensorcnt), one 16x128 f32 tile per K-half wave.
// ---------------------------------------------------------------------------

typedef __bf16 bf16;
typedef __attribute__((ext_vector_type(8)))  __bf16 v8bf;
typedef __attribute__((ext_vector_type(16))) __bf16 v16bf;
typedef __attribute__((ext_vector_type(8)))  float  v8f;
typedef __attribute__((ext_vector_type(4)))  float  v4f;
typedef __attribute__((ext_vector_type(4)))  unsigned int v4u;
typedef __attribute__((ext_vector_type(8)))  int v8i;
typedef __attribute__((ext_vector_type(4)))  int v4i;

__device__ __forceinline__ v16bf cat8(v8bf lo, v8bf hi) {
  return __builtin_shufflevector(lo, hi, 0,1,2,3,4,5,6,7,8,9,10,11,12,13,14,15);
}

__device__ __forceinline__ v8f wmma_bf16(v16bf a, v16bf b, v8f c) {
  // D = A(16x32 bf16) * B(32x16 bf16) + C(16x16 f32)
  return __builtin_amdgcn_wmma_f32_16x16x32_bf16(false, a, false, b,
                                                 (short)0, c, false, false);
}

// A-fragment (16x32 bf16) from row-major bf16 [rows][pitch].
// Lane L holds row (lane&15); K-offsets {0..7,16..23} (+8 for lanes 16..31).
__device__ __forceinline__ v16bf load_a_bf16(const bf16* base, long pitch,
                                             int kbase, int lane) {
  const bf16* p = base + (long)(lane & 15) * pitch + kbase + ((lane & 16) ? 8 : 0);
  v8bf lo = *(const v8bf*)p;
  v8bf hi = *(const v8bf*)(p + 16);
  return cat8(lo, hi);
}

// A-fragment from row-major f32 (converted to bf16 in-register).
__device__ __forceinline__ v16bf load_a_f32(const float* base, long pitch,
                                            int row, int kbase, int lane) {
  const float* p = base + (long)row * pitch + kbase + ((lane & 16) ? 8 : 0);
  v4f a0 = *(const v4f*)p;
  v4f a1 = *(const v4f*)(p + 4);
  v4f b0 = *(const v4f*)(p + 16);
  v4f b1 = *(const v4f*)(p + 20);
  v16bf r;
#pragma unroll
  for (int i = 0; i < 4; ++i) {
    r[i]      = (bf16)a0[i];
    r[4 + i]  = (bf16)a1[i];
    r[8 + i]  = (bf16)b0[i];
    r[12 + i] = (bf16)b1[i];
  }
  return r;
}

// B-fragment (32x16 bf16) from "NT" weights: memory [Ncols][K] row-major
// (nn.Linear layout). Lane L holds output-column (lane&15);
// K-offsets {0..7,16..23} (+8 for lanes 16..31), matching the A-fragment.
__device__ __forceinline__ v16bf load_b_bf16(const bf16* base, long pitch,
                                             int kbase, int lane) {
  const bf16* p = base + (long)(lane & 15) * pitch + kbase + ((lane & 16) ? 8 : 0);
  v8bf lo = *(const v8bf*)p;
  v8bf hi = *(const v8bf*)(p + 16);
  return cat8(lo, hi);
}

// B-fragment from f32 memory laid out [K][pitch] (K along rows): strided.
__device__ __forceinline__ v16bf load_b_f32_rowK(const float* base, long pitch,
                                                 int col, int kbase, int lane) {
  int k0 = kbase + ((lane & 16) ? 8 : 0);
  v16bf r;
#pragma unroll
  for (int j = 0; j < 16; ++j)
    r[j] = (bf16)base[(long)(k0 + j + (j >= 8 ? 8 : 0)) * pitch + col];
  return r;
}

// ---------------------------------------------------------------------------
// Tensor Data Mover: 2-D tile load (rows x 128 f32, row stride 128) from
// global to LDS.  D# built per CDNA5 ISA Sect. 8:
//  group0: count=1, lds_addr, global_addr[56:0], type=2
//  group1: data_size=4B, tensor_dim0=128, tensor_dim1=rows,
//          tile_dim0=128, tile_dim1=rows, tensor_dim0_stride=128
//  groups 2/3 (+spare group): zero (2-D tensor, no iterate/gather)
// Toolchain uses the 6-arg builtin form (g0, g1, g2, g3, g4, cpol).
// ---------------------------------------------------------------------------
__device__ __forceinline__ void tdm_load_rows_f32(const float* gsrc,
                                                  unsigned lds_byte_off,
                                                  unsigned rows) {
  unsigned long long ga = (unsigned long long)(const void*)gsrc;
  v4u g0;
  g0[0] = 1u;                                    // count=1 (valid), user mode
  g0[1] = lds_byte_off;                          // lds_addr
  g0[2] = (unsigned)(ga & 0xffffffffu);          // global_addr[31:0]
  g0[3] = (unsigned)((ga >> 32) & 0x01ffffffu)   // global_addr[56:32]
          | (2u << 30);                          // type=2 ("image")
  v8i g1;
  g1[0] = (int)(2u << 16);                       // data_size=2 (4 bytes)
  g1[1] = (int)(128u << 16);                     // tensor_dim0 = 128
  g1[2] = (int)((rows & 0xffffu) << 16);         // tensor_dim1 = rows
  g1[3] = (int)(128u << 16);                     // tile_dim0 = 128
  g1[4] = (int)(rows & 0xffffu);                 // tile_dim1 = rows
  g1[5] = 128;                                   // tensor_dim0_stride = 128
  g1[6] = 0;
  g1[7] = 0;
  v4i z4 = {0, 0, 0, 0};
  v8i z8 = {0, 0, 0, 0, 0, 0, 0, 0};
  __builtin_amdgcn_tensor_load_to_lds(g0, g1, z4, z4, z8, 0);
}

// ---------------------------------------------------------------------------
// Kernel W: one-time weight convert to bf16 (+ W1 transpose -> W1T[x][d]).
// ---------------------------------------------------------------------------
__global__ __launch_bounds__(256) void kw_wprep(
    const float* __restrict__ W1, const float* __restrict__ W2,
    const float* __restrict__ Wv, const float* __restrict__ Wf,
    const float* __restrict__ g1w, const float* __restrict__ g2w,
    bf16* __restrict__ W1T_b, bf16* __restrict__ W2_b, bf16* __restrict__ Wv_b,
    bf16* __restrict__ Wf_b, bf16* __restrict__ G1_b, bf16* __restrict__ G2_b) {
  long i = (long)blockIdx.x * 256 + threadIdx.x;
  long stride = (long)gridDim.x * 256;
  for (long t = i; t < 1048576; t += stride) W2_b[t] = (bf16)W2[t];
  for (long t = i; t < 1179648; t += stride) Wv_b[t] = (bf16)Wv[t];
  for (long t = i; t < 262144;  t += stride) Wf_b[t] = (bf16)Wf[t];
  for (long t = i; t < 1179648; t += stride) G1_b[t] = (bf16)g1w[t];
  for (long t = i; t < 1048576; t += stride) G2_b[t] = (bf16)g2w[t];
  for (long t = i; t < 1179648; t += stride) {   // W1[d][x] -> W1T[x][d]
    long d = t / 1152, x = t - d * 1152;
    W1T_b[x * 1024 + d] = (bf16)W1[t];
  }
}

// ---------------------------------------------------------------------------
// Kernel A: per 16-chunk tile, batched WMMA GEMM chain:
//   gin=[w|mean_k Et], fCb=bf16(fC)
//   fCp = fC@Wf^T ; Q = gin[:,:1024]@W2^T + beta*fCp -> Qb(bf16)
//   Qt_h = Q_h @ W1_h  (via pre-transposed W1T, NT fragments)
//   hmid = relu(gin@g1^T + b1) ; g = sigmoid(hmid@g2^T + b2)
// grid=128 (C/16), block=256 (8 waves). All staging via ws (L2-resident).
// ---------------------------------------------------------------------------
__global__ __launch_bounds__(256) void ka_prep(
    const float* __restrict__ w, const float* __restrict__ Et,
    const float* __restrict__ fC, const float* __restrict__ g1b,
    const float* __restrict__ g2b, const float* __restrict__ beta_p,
    const bf16* __restrict__ Wf_b, const bf16* __restrict__ W2_b,
    const bf16* __restrict__ W1T_b, const bf16* __restrict__ G1_b,
    const bf16* __restrict__ G2_b,
    bf16* __restrict__ gin_ws, bf16* __restrict__ fCb_ws,
    bf16* __restrict__ Qb_ws, bf16* __restrict__ hb_ws,
    float* __restrict__ fCp_ws, float* __restrict__ g_ws,
    bf16* __restrict__ Qt_ws) {
  const int tid = threadIdx.x;
  const int lane = tid & 31;
  const int wv = tid >> 5;
  const int cbase = blockIdx.x * 16;
  const float beta = beta_p[0];

  // ---- phase 0: activations -> bf16 ws ----
  for (int i = tid; i < 16 * 1024; i += 256) {
    int c = i >> 10, d = i & 1023;
    gin_ws[(long)(cbase + c) * 1152 + d] = (bf16)w[(long)(cbase + c) * 1024 + d];
  }
  for (int i = tid; i < 16 * 128; i += 256) {
    int c = i >> 7, e = i & 127;
    const float* p = Et + (long)(cbase + c) * 64 * 128 + e;
    float s = 0.f;
#pragma unroll 8
    for (int k = 0; k < 64; ++k) s += p[k * 128];
    gin_ws[(long)(cbase + c) * 1152 + 1024 + e] = (bf16)(s * (1.0f / 64.0f));
  }
  for (int i = tid; i < 16 * 256; i += 256) {
    int c = i >> 8, e = i & 255;
    fCb_ws[(long)(cbase + c) * 256 + e] = (bf16)fC[(long)(cbase + c) * 256 + e];
  }
  // zero fake heads 8..15 of Qt (B-fragment lanes 8..15 read them)
  for (int i = tid; i < 16 * 8 * 1152; i += 256) {
    int c = i / (8 * 1152), rem = i - c * 8 * 1152;
    Qt_ws[((long)(cbase + c) * 16 + 8) * 1152 + rem] = (bf16)0.f;
  }
  __threadfence();
  __syncthreads();

  const bf16* Agin = gin_ws + (long)cbase * 1152;
  const bf16* AfC  = fCb_ws + (long)cbase * 256;
  const int n = lane & 15, rh = (lane >> 4) * 8;

  // ---- fCp = fC @ Wf^T  (N=1024, K=256) ----
  for (int i = 0; i < 8; ++i) {
    int nt = wv * 8 + i;
    v8f acc = {};
    for (int kt = 0; kt < 8; ++kt)
      acc = wmma_bf16(load_a_bf16(AfC, 256, kt * 32, lane),
                      load_b_bf16(Wf_b + (long)nt * 16 * 256, 256, kt * 32, lane), acc);
#pragma unroll
    for (int r = 0; r < 8; ++r)
      fCp_ws[(long)(cbase + r + rh) * 1024 + nt * 16 + n] = acc[r];
  }
  __threadfence();
  __syncthreads();

  // ---- Q = w @ W2^T + beta*fCp  -> Qb (bf16) (N=1024, K=1024) ----
  for (int i = 0; i < 8; ++i) {
    int nt = wv * 8 + i;
    v8f acc = {};
    for (int kt = 0; kt < 32; ++kt)
      acc = wmma_bf16(load_a_bf16(Agin, 1152, kt * 32, lane),
                      load_b_bf16(W2_b + (long)nt * 16 * 1024, 1024, kt * 32, lane), acc);
#pragma unroll
    for (int r = 0; r < 8; ++r) {
      long idx = (long)(cbase + r + rh) * 1024 + nt * 16 + n;
      Qb_ws[idx] = (bf16)(acc[r] + beta * fCp_ws[idx]);
    }
  }
  __threadfence();
  __syncthreads();

  // ---- Qt_h = Q_h @ W1_h  via W1T (N=1152 per head, K=128) ----
  for (int h = 0; h < 8; ++h) {
    const bf16* Aq = Qb_ws + (long)cbase * 1024 + h * 128;
    for (int i = 0; i < 9; ++i) {
      int nt = wv * 9 + i;
      v8f acc = {};
#pragma unroll
      for (int kt = 0; kt < 4; ++kt)
        acc = wmma_bf16(load_a_bf16(Aq, 1024, kt * 32, lane),
                        load_b_bf16(W1T_b + (long)nt * 16 * 1024 + h * 128,
                                    1024, kt * 32, lane), acc);
#pragma unroll
      for (int r = 0; r < 8; ++r)
        Qt_ws[((long)(cbase + r + rh) * 16 + h) * 1152 + nt * 16 + n] = (bf16)acc[r];
    }
  }

  // ---- hmid = relu(gin @ g1^T + b1) (N=1024, K=1152) ----
  for (int i = 0; i < 8; ++i) {
    int nt = wv * 8 + i;
    v8f acc = {};
    for (int kt = 0; kt < 36; ++kt)
      acc = wmma_bf16(load_a_bf16(Agin, 1152, kt * 32, lane),
                      load_b_bf16(G1_b + (long)nt * 16 * 1152, 1152, kt * 32, lane), acc);
    int d = nt * 16 + n;
    float bias = g1b[d];
#pragma unroll
    for (int r = 0; r < 8; ++r)
      hb_ws[(long)(cbase + r + rh) * 1024 + d] = (bf16)fmaxf(acc[r] + bias, 0.f);
  }
  __threadfence();
  __syncthreads();

  // ---- g = sigmoid(hmid @ g2^T + b2) (N=1024, K=1024) ----
  for (int i = 0; i < 8; ++i) {
    int nt = wv * 8 + i;
    v8f acc = {};
    for (int kt = 0; kt < 32; ++kt)
      acc = wmma_bf16(load_a_bf16(hb_ws + (long)cbase * 1024, 1024, kt * 32, lane),
                      load_b_bf16(G2_b + (long)nt * 16 * 1024, 1024, kt * 32, lane), acc);
    int d = nt * 16 + n;
    float bias = g2b[d];
#pragma unroll
    for (int r = 0; r < 8; ++r) {
      float v = acc[r] + bias;
      g_ws[(long)(cbase + r + rh) * 1024 + d] = 1.0f / (1.0f + __expf(-v));
    }
  }
}

// ---------------------------------------------------------------------------
// Kernel B: per-chunk attention. Streams F (512 MB) straight into WMMA
// fragments; Et[c] (32 KB) is TDM-DMA'd into LDS (one 16-row tile per K-half
// wave, overlapped, completion via s_wait_tensorcnt). scores D[k,h] register
// layout == attn A-fragment layout for the aF WMMA, so softmax never leaves
// registers (one shfl_xor(16) per reduce).
// grid=2048, block=256 (8 waves: 4 k-Mtiles x 2 K-halves).
// Note: alpha*fCp shift on K is constant over k -> cancels in softmax.
// ---------------------------------------------------------------------------
__global__ __launch_bounds__(256) void kb_attn(
    const float* __restrict__ F, const float* __restrict__ Et,
    const bf16* __restrict__ Qt_ws, bf16* __restrict__ aF_ws) {
  __shared__ float scores_s[2][64][16];
  __shared__ float et_lds[64 * 128];  // Et[c] staged by the Tensor Data Mover
  const int tid = threadIdx.x, lane = tid & 31, wv = tid >> 5;
  const long c = blockIdx.x;
  const int m = wv & 3, half = wv >> 2;
  const float* Fc = F + c * 64 * 1024;
  const float* Ec = Et + c * 64 * 128;
  const bf16*  Qc = Qt_ws + c * 16 * 1152;

  // Waves 4..7 (K-half 1, which consumes Et columns) each DMA their 16 rows
  // of Et[c] into LDS and wait on TENSORcnt before first use.
  if (half == 1) {
    unsigned lds_off = (unsigned)(unsigned long long)(const void*)&et_lds[m * 16 * 128];
    tdm_load_rows_f32(Ec + (long)m * 16 * 128, lds_off, 16u);
    __builtin_amdgcn_s_wait_tensorcnt(0);
  }

  // ---- scores partial: D[16k x 16h] for Mtile m, K-half ----
  v8f acc = {};
  const int arow = m * 16 + (lane & 15);
  for (int kt = half * 18; kt < half * 18 + 18; ++kt) {
    if (kt < 31)
      __builtin_prefetch(Fc + (long)arow * 1024 + (kt + 1) * 32, 0, 1);
    v16bf a = (kt < 32) ? load_a_f32(Fc, 1024, arow, kt * 32, lane)
                        : load_a_f32(et_lds, 128, arow, kt * 32 - 1024, lane);
    v16bf b = load_b_bf16(Qc, 1152, kt * 32, lane);
    acc = wmma_bf16(a, b, acc);
  }
  {
    int n = lane & 15, rh = (lane >> 4) * 8;
#pragma unroll
    for (int r = 0; r < 8; ++r) scores_s[half][m * 16 + r + rh][n] = acc[r];
  }
  __syncthreads();   // also publishes TDM-written et_lds to all waves

  // ---- softmax over k (each wave redundantly; result = A-fragments) ----
  const float scale = 0.088388347648318447f;  // 1/sqrt(128)
  const int h = lane & 15, rh = (lane >> 4) * 8;
  float sc[4][8];
  float mx = -3.0e38f;
#pragma unroll
  for (int mi = 0; mi < 4; ++mi)
#pragma unroll
    for (int r = 0; r < 8; ++r) {
      int k = mi * 16 + r + rh;
      float s = (scores_s[0][k][h] + scores_s[1][k][h]) * scale;
      sc[mi][r] = s;
      mx = fmaxf(mx, s);
    }
  mx = fmaxf(mx, __shfl_xor(mx, 16, 32));
  float sum = 0.f;
#pragma unroll
  for (int mi = 0; mi < 4; ++mi)
#pragma unroll
    for (int r = 0; r < 8; ++r) {
      float e = __expf(sc[mi][r] - mx);
      sc[mi][r] = e;
      sum += e;
    }
  sum += __shfl_xor(sum, 16, 32);
  const float inv = 1.0f / sum;

  v16bf af0, af1;  // attn A-fragments for K-tiles k=0..31 and k=32..63
#pragma unroll
  for (int r = 0; r < 8; ++r) {
    af0[r]     = (bf16)(sc[0][r] * inv);
    af0[8 + r] = (bf16)(sc[1][r] * inv);
    af1[r]     = (bf16)(sc[2][r] * inv);
    af1[8 + r] = (bf16)(sc[3][r] * inv);
  }

  // ---- aF[h,x] = attn @ Fp : D[16h x 16x], 72 N-tiles split over waves ----
  for (int i = 0; i < 9; ++i) {
    int nt = wv + 8 * i;
    v16bf b0, b1;
    if (nt < 64) {
      int col = nt * 16 + (lane & 15);
      b0 = load_b_f32_rowK(Fc, 1024, col, 0, lane);
      b1 = load_b_f32_rowK(Fc, 1024, col, 32, lane);
    } else {
      int col = (nt - 64) * 16 + (lane & 15);
      b0 = load_b_f32_rowK(et_lds, 128, col, 0, lane);
      b1 = load_b_f32_rowK(et_lds, 128, col, 32, lane);
    }
    v8f o = {};
    o = wmma_bf16(af0, b0, o);
    o = wmma_bf16(af1, b1, o);
    if (lane < 16) {  // rows h=0..7 live in lanes 0..15
#pragma unroll
      for (int r = 0; r < 8; ++r)
        aF_ws[(c * 8 + r) * 1152 + nt * 16 + lane] = (bf16)o[r];
    }
  }
}

// ---------------------------------------------------------------------------
// Kernel C: A = aF @ Wv^T (+alpha*fCp), gated residual, L2-normalize rows.
// grid=128 (16 chunks), block=256 (wave w == head w).
// ---------------------------------------------------------------------------
__global__ __launch_bounds__(256) void kc_out(
    const float* __restrict__ w, const bf16* __restrict__ aF_ws,
    const bf16* __restrict__ Wv_b, const float* __restrict__ fCp_ws,
    const float* __restrict__ g_ws, const float* __restrict__ alpha_p,
    float* __restrict__ outp) {
  __shared__ float ssq[16];
  const int tid = threadIdx.x, lane = tid & 31, wv = tid >> 5;
  const int cbase = blockIdx.x * 16;
  const float alpha = alpha_p[0];
  if (tid < 16) ssq[tid] = 0.f;
  __syncthreads();

  const int h = wv;  // 8 waves == 8 heads
  const bf16* Abase = aF_ws + ((long)cbase * 8 + h) * 1152;  // row stride 8*1152
  const int n = lane & 15, rh = (lane >> 4) * 8;
  float part[8];
#pragma unroll
  for (int r = 0; r < 8; ++r) part[r] = 0.f;

  for (int i = 0; i < 8; ++i) {
    const int d0 = h * 128 + i * 16;
    v8f acc = {};
    for (int kt = 0; kt < 36; ++kt)
      acc = wmma_bf16(load_a_bf16(Abase, 8 * 1152, kt * 32, lane),
                      load_b_bf16(Wv_b + (long)d0 * 1152, 1152, kt * 32, lane), acc);
    const int d = d0 + n;
#pragma unroll
    for (int r = 0; r < 8; ++r) {
      long row = cbase + r + rh;
      float A = acc[r] + alpha * fCp_ws[row * 1024 + d];
      float pre = w[row * 1024 + d] + g_ws[row * 1024 + d] * A;
      outp[row * 1024 + d] = pre;
      part[r] += pre * pre;
    }
  }
#pragma unroll
  for (int r = 0; r < 8; ++r) atomicAdd(&ssq[r + rh], part[r]);
  __threadfence();
  __syncthreads();

  for (int i = tid; i < 16 * 1024; i += 256) {
    int cl = i >> 10, d = i & 1023;
    float s = 1.0f / fmaxf(sqrtf(ssq[cl]), 1e-12f);
    outp[(long)(cbase + cl) * 1024 + d] *= s;
  }
}

// ---------------------------------------------------------------------------
extern "C" void kernel_launch(void* const* d_in, const int* in_sizes, int n_in,
                              void* d_out, int out_size, void* d_ws, size_t ws_size,
                              hipStream_t stream) {
  const float* w     = (const float*)d_in[0];
  const float* F     = (const float*)d_in[1];
  const float* Et    = (const float*)d_in[2];
  const float* fC    = (const float*)d_in[3];
  const float* W1    = (const float*)d_in[4];
  const float* W2    = (const float*)d_in[5];
  const float* Wv    = (const float*)d_in[6];
  const float* Wf    = (const float*)d_in[7];
  const float* alpha = (const float*)d_in[8];
  const float* beta  = (const float*)d_in[9];
  const float* g1w   = (const float*)d_in[10];
  const float* g1b   = (const float*)d_in[11];
  const float* g2w   = (const float*)d_in[12];
  const float* g2b   = (const float*)d_in[13];
  float* outp = (float*)d_out;

  char* base = (char*)d_ws;
  size_t off = 0;
  auto take = [&](size_t bytes) -> char* {
    char* p = base + off;
    off = (off + bytes + 255) & ~(size_t)255;
    return p;
  };
  bf16* W1T_b  = (bf16*)take(1179648u * 2);
  bf16* W2_b   = (bf16*)take(1048576u * 2);
  bf16* Wv_b   = (bf16*)take(1179648u * 2);
  bf16* Wf_b   = (bf16*)take(262144u * 2);
  bf16* G1_b   = (bf16*)take(1179648u * 2);
  bf16* G2_b   = (bf16*)take(1048576u * 2);
  bf16* gin_ws = (bf16*)take((size_t)2048 * 1152 * 2);
  bf16* fCb_ws = (bf16*)take((size_t)2048 * 256 * 2);
  bf16* Qb_ws  = (bf16*)take((size_t)2048 * 1024 * 2);
  bf16* hb_ws  = (bf16*)take((size_t)2048 * 1024 * 2);
  bf16* Qt_ws  = (bf16*)take((size_t)2048 * 16 * 1152 * 2);
  bf16* aF_ws  = (bf16*)take((size_t)2048 * 8 * 1152 * 2);
  float* fCp_ws = (float*)take((size_t)2048 * 1024 * 4);
  float* g_ws   = (float*)take((size_t)2048 * 1024 * 4);

  kw_wprep<<<1024, 256, 0, stream>>>(W1, W2, Wv, Wf, g1w, g2w,
                                     W1T_b, W2_b, Wv_b, Wf_b, G1_b, G2_b);
  ka_prep<<<128, 256, 0, stream>>>(w, Et, fC, g1b, g2b, beta,
                                   Wf_b, W2_b, W1T_b, G1_b, G2_b,
                                   gin_ws, fCb_ws, Qb_ws, hb_ws,
                                   fCp_ws, g_ws, Qt_ws);
  kb_attn<<<2048, 256, 0, stream>>>(F, Et, Qt_ws, aF_ws);
  kc_out<<<128, 256, 0, stream>>>(w, aF_ws, Wv_b, fCp_ws, g_ws, alpha, outp);
}
